// MultiHeadSelfAttention_18700287606946
// MI455X (gfx1250) — matrix operand order repaired and checked by reading, not compiled, not run on hardware
//
#include <hip/hip_runtime.h>
#include <hip/hip_bf16.h>
#include <stdint.h>

#define DIM   1024
#define NHEAD 16
#define HDIM  64
#define BATCH 2
#define SEQ   2048
#define TOK   (BATCH*SEQ)   // 4096

typedef __bf16 bf16_t;
typedef __attribute__((ext_vector_type(16))) __bf16        v16bf;
typedef __attribute__((ext_vector_type(8)))  float         v8f;
typedef __attribute__((ext_vector_type(8)))  unsigned int  v8u;

// Load a 32-byte WMMA fragment as two 16-byte chunks (global or LDS).
__device__ __forceinline__ v16bf load_frag(const bf16_t* p0, const bf16_t* p1) {
    uint4 a = *reinterpret_cast<const uint4*>(p0);
    uint4 b = *reinterpret_cast<const uint4*>(p1);
    v8u r = { a.x, a.y, a.z, a.w, b.x, b.y, b.z, b.w };
    return __builtin_bit_cast(v16bf, r);
}

__device__ __forceinline__ v8f wmma_bf16(v16bf a, v16bf b, v8f c) {
    // (neg_a, A, neg_b, B, c_mod, C, reuse_a, reuse_b)
    return __builtin_amdgcn_wmma_f32_16x16x32_bf16(false, a, false, b, (short)0, c, false, false);
}

// ---------------------------------------------------------------------------
// Elementwise fp32 -> bf16
__global__ void f32_to_bf16_kernel(const float* __restrict__ in, bf16_t* __restrict__ out, int n) {
    int i = blockIdx.x * 256 + threadIdx.x;
    if (i < n) out[i] = (bf16_t)in[i];
}

// W (K x N, row-major, f32) -> WT (N x K, row-major, bf16). LDS-tiled, coalesced
// on both sides; 33-stride padding avoids bank conflicts on the 64-bank LDS.
__global__ __launch_bounds__(256) void transpose_w_kernel(const float* __restrict__ W,
                                                          bf16_t* __restrict__ WT,
                                                          int K, int N) {
    __shared__ float tile[32][33];
    const int k0 = blockIdx.x * 32;
    const int n0 = blockIdx.y * 32;
    const int tx = threadIdx.x;          // 32
    const int ty = threadIdx.y;          // 8
#pragma unroll
    for (int j = 0; j < 4; ++j)
        tile[ty + j * 8][tx] = W[(size_t)(k0 + ty + j * 8) * N + n0 + tx];
    __syncthreads();
#pragma unroll
    for (int j = 0; j < 4; ++j)
        WT[(size_t)(n0 + ty + j * 8) * K + k0 + tx] = (bf16_t)tile[tx][ty + j * 8];
}

// ---------------------------------------------------------------------------
// GEMM: C(MxN, f32) = A(MxK, bf16) @ BT(NxK, bf16)^T + bias
// Block = 128 threads = 4 waves; each wave computes a 32x64 tile (2 m-tiles x
// 4 n-tiles, 8 fp32 accumulators). K-loop step 64 with ping-pong fragment
// buffers so the next chunk's loads overlap the current chunk's 8 WMMAs.
__global__ __launch_bounds__(128) void gemm_bf16_kernel(
    const bf16_t* __restrict__ A, const bf16_t* __restrict__ BT,
    const float* __restrict__ bias, float* __restrict__ C,
    int M, int N, int K)
{
    const int lane = threadIdx.x & 31;
    const int wave = threadIdx.x >> 5;
    const int lrow = lane & 15;
    const int lhal = lane >> 4;
    const int m0 = (blockIdx.x * 8 + wave * 2) * 16;
    const int n0 = blockIdx.y * 64;
    if (m0 >= M) return;

    v8f acc[8] = {};
    const bf16_t* a0 = A + (size_t)(m0 + lrow) * K + lhal * 8;   // A-frag: chunks @K, K+16
    const bf16_t* a1 = a0 + (size_t)16 * K;
    const bf16_t* b0 = BT + (size_t)(n0 + lrow) * K + lhal * 8;  // B-frag: chunks @K, K+16

    v16bf afA[2], bfA[4], afB[2], bfB[4];
    // prologue: chunk 0
    afA[0] = load_frag(a0, a0 + 16);
    afA[1] = load_frag(a1, a1 + 16);
#pragma unroll
    for (int nt = 0; nt < 4; ++nt) {
        const bf16_t* bp = b0 + (size_t)nt * 16 * K;
        bfA[nt] = load_frag(bp, bp + 16);
    }

    for (int kk = 0; kk < K; kk += 64) {
        {   // load odd chunk (kk+32) while A-buffer computes
            const int k2 = kk + 32;
            afB[0] = load_frag(a0 + k2, a0 + k2 + 16);
            afB[1] = load_frag(a1 + k2, a1 + k2 + 16);
#pragma unroll
            for (int nt = 0; nt < 4; ++nt) {
                const bf16_t* bp = b0 + (size_t)nt * 16 * K + k2;
                bfB[nt] = load_frag(bp, bp + 16);
            }
        }
#pragma unroll
        for (int mt = 0; mt < 2; ++mt)
#pragma unroll
            for (int nt = 0; nt < 4; ++nt)
                acc[mt * 4 + nt] = wmma_bf16(afA[mt], bfA[nt], acc[mt * 4 + nt]);

        if (kk + 64 < K) {   // load next even chunk while B-buffer computes
            const int k2 = kk + 64;
            afA[0] = load_frag(a0 + k2, a0 + k2 + 16);
            afA[1] = load_frag(a1 + k2, a1 + k2 + 16);
#pragma unroll
            for (int nt = 0; nt < 4; ++nt) {
                const bf16_t* bp = b0 + (size_t)nt * 16 * K + k2;
                bfA[nt] = load_frag(bp, bp + 16);
            }
        }
#pragma unroll
        for (int mt = 0; mt < 2; ++mt)
#pragma unroll
            for (int nt = 0; nt < 4; ++nt)
                acc[mt * 4 + nt] = wmma_bf16(afB[mt], bfB[nt], acc[mt * 4 + nt]);
    }

#pragma unroll
    for (int mt = 0; mt < 2; ++mt)
#pragma unroll
        for (int nt = 0; nt < 4; ++nt) {
            int col = n0 + nt * 16 + lrow;
            float bv = bias ? bias[col] : 0.0f;
#pragma unroll
            for (int r = 0; r < 8; ++r) {
                int row = m0 + mt * 16 + lhal * 8 + r;
                C[(size_t)row * N + col] = acc[mt * 4 + nt][r] + bv;
            }
        }
}

// ---------------------------------------------------------------------------
// RoPE + head split for Q,K only: qkv f32 [B,T,3*DIM] -> Q,K [B,H,T,64] bf16.
__global__ void rope_qk_kernel(const float* __restrict__ qkv,
                               const float* __restrict__ sin_, const float* __restrict__ cos_,
                               bf16_t* __restrict__ Q, bf16_t* __restrict__ Kp)
{
    int idx = blockIdx.x * 256 + threadIdx.x;     // over TOK*DIM = 2^22
    if (idx >= TOK * DIM) return;
    int d = idx & 63;
    int h = (idx >> 6) & 15;
    int t = (idx >> 10) & (SEQ - 1);
    int b = idx >> 21;

    size_t base = (size_t)(b * SEQ + t) * (3 * DIM) + h * HDIM + d;
    float qv = qkv[base];
    float kv = qkv[base + DIM];
    float s = sin_[t * HDIM + d];
    float c = cos_[t * HDIM + d];
    size_t rot = (d < 32) ? (base + 32) : (base - 32);
    float qr = (d < 32) ? -qkv[rot]       :  qkv[rot];
    float kr = (d < 32) ? -qkv[rot + DIM] :  qkv[rot + DIM];

    size_t qidx = ((size_t)(b * NHEAD + h) * SEQ + t) * HDIM + d;
    Q[qidx]  = (bf16_t)(qv * c + qr * s);
    Kp[qidx] = (bf16_t)(kv * c + kr * s);
}

// V slice of qkv -> V^T [B,H,64,T] bf16, LDS-tiled so both sides are coalesced.
__global__ __launch_bounds__(256) void v_transpose_kernel(const float* __restrict__ qkv,
                                                          bf16_t* __restrict__ VT)
{
    __shared__ float tile[32][65];       // 65 stride: conflict-free column reads
    const int t0 = blockIdx.x * 32;
    const int bh = blockIdx.y;
    const int b = bh >> 4, h = bh & 15;
    const int tid = threadIdx.x;         // 256
#pragma unroll
    for (int i = 0; i < 8; ++i) {
        int l = tid + i * 256;           // 32 t x 64 d tile
        int tl = l >> 6, d = l & 63;
        tile[tl][d] = qkv[(size_t)(b * SEQ + t0 + tl) * (3 * DIM) + 2 * DIM + h * HDIM + d];
    }
    __syncthreads();
#pragma unroll
    for (int i = 0; i < 8; ++i) {
        int l = tid + i * 256;
        int tl = l & 31, d = l >> 5;
        VT[((size_t)bh * HDIM + d) * SEQ + t0 + tl] = (bf16_t)tile[tl][d];
    }
}

// ---------------------------------------------------------------------------
// Flash-style causal attention: 1 wave per (bh, 16-row q-tile).
// K-fragments for block kb+1 and V-fragments for block kb are loaded before
// the softmax VALU chain so VMEM overlaps the exp/shuffle work.
__global__ __launch_bounds__(32) void attn_kernel(
    const bf16_t* __restrict__ Q, const bf16_t* __restrict__ Kt,
    const bf16_t* __restrict__ VT, bf16_t* __restrict__ Y)
{
    __shared__ bf16_t Plds[16 * 32];
    const int lane = threadIdx.x;
    const int lrow = lane & 15;
    const int lhal = lane >> 4;
    const int qt = blockIdx.x;
    const int bh = blockIdx.y;
    const int b  = bh >> 4;
    const int h  = bh & 15;
    const int q0 = qt * 16;

    const bf16_t* Qb = Q  + (size_t)bh * SEQ * HDIM;
    const bf16_t* Kb = Kt + (size_t)bh * SEQ * HDIM;
    const bf16_t* Vb = VT + (size_t)bh * HDIM * SEQ;

    v16bf qf[2];
    {
        const bf16_t* qrow = Qb + (size_t)(q0 + lrow) * HDIM + lhal * 16;
        qf[0] = load_frag(qrow,      qrow + 8);
        qf[1] = load_frag(qrow + 32, qrow + 40);
    }

    v8f acc[4] = {};
    float Mrow[8], Lrow[8];
#pragma unroll
    for (int r = 0; r < 8; ++r) { Mrow[r] = -1e30f; Lrow[r] = 0.0f; }

    const int nkb = (q0 + 16 + 31) >> 5;

    // preload K-fragments for block 0: kf[2*t2+j] = keys tile t2, d half j
    v16bf kf[4];
#pragma unroll
    for (int t2 = 0; t2 < 2; ++t2) {
        const bf16_t* krow = Kb + (size_t)(t2 * 16 + lrow) * HDIM + lhal * 16;
        kf[2 * t2]     = load_frag(krow,      krow + 8);
        kf[2 * t2 + 1] = load_frag(krow + 32, krow + 40);
    }

    for (int kb = 0; kb < nkb; ++kb) {
        const int k0 = kb * 32;

        v8f s[2] = {};
#pragma unroll
        for (int t2 = 0; t2 < 2; ++t2) {
            s[t2] = wmma_bf16(qf[0], kf[2 * t2],     s[t2]);
            s[t2] = wmma_bf16(qf[1], kf[2 * t2 + 1], s[t2]);
        }

        // issue next block's K loads + this block's V loads before softmax VALU
        v16bf kn[4];
        if (kb + 1 < nkb) {
            const int k1 = k0 + 32;
#pragma unroll
            for (int t2 = 0; t2 < 2; ++t2) {
                const bf16_t* krow = Kb + (size_t)(k1 + t2 * 16 + lrow) * HDIM + lhal * 16;
                kn[2 * t2]     = load_frag(krow,      krow + 8);
                kn[2 * t2 + 1] = load_frag(krow + 32, krow + 40);
            }
        }
        v16bf vf[4];
#pragma unroll
        for (int nt = 0; nt < 4; ++nt) {
            const bf16_t* vrow = Vb + (size_t)(nt * 16 + lrow) * SEQ + k0 + lhal * 16;
            vf[nt] = load_frag(vrow, vrow + 8);
        }

        const float scale = 0.125f;     // 1/sqrt(64)
#pragma unroll
        for (int r = 0; r < 8; ++r) {
            int qg   = q0 + lhal * 8 + r;
            float s0 = (k0 + lrow      <= qg) ? s[0][r] * scale : -1e30f;
            float s1 = (k0 + 16 + lrow <= qg) ? s[1][r] * scale : -1e30f;
            float mx = fmaxf(s0, s1);
#pragma unroll
            for (int m = 1; m < 16; m <<= 1) mx = fmaxf(mx, __shfl_xor(mx, m, 32));
            float Mn = fmaxf(Mrow[r], mx);
            float al = __expf(Mrow[r] - Mn);
            float p0 = __expf(s0 - Mn);
            float p1 = __expf(s1 - Mn);
            float rs = p0 + p1;
#pragma unroll
            for (int m = 1; m < 16; m <<= 1) rs += __shfl_xor(rs, m, 32);
            Lrow[r] = Lrow[r] * al + rs;
            Mrow[r] = Mn;
#pragma unroll
            for (int nt = 0; nt < 4; ++nt) acc[nt][r] *= al;
            int mr = lhal * 8 + r;                 // C-layout row -> row-major P in LDS
            Plds[mr * 32 + lrow]      = (bf16_t)p0;
            Plds[mr * 32 + 16 + lrow] = (bf16_t)p1;
        }
        __syncthreads();

        const bf16_t* prow = &Plds[lrow * 32 + lhal * 16];
        v16bf pf = load_frag(prow, prow + 8);      // P as 16x32 A-fragment
#pragma unroll
        for (int nt = 0; nt < 4; ++nt)
            acc[nt] = wmma_bf16(pf, vf[nt], acc[nt]);
        __syncthreads();

#pragma unroll
        for (int j = 0; j < 4; ++j) kf[j] = kn[j];
    }

#pragma unroll
    for (int nt = 0; nt < 4; ++nt) {
        int d = nt * 16 + lrow;
#pragma unroll
        for (int r = 0; r < 8; ++r) {
            int tq = q0 + lhal * 8 + r;
            float y = acc[nt][r] / Lrow[r];
            Y[(size_t)(b * SEQ + tq) * DIM + h * HDIM + d] = (bf16_t)y;
        }
    }
}

// ---------------------------------------------------------------------------
extern "C" void kernel_launch(void* const* d_in, const int* in_sizes, int n_in,
                              void* d_out, int out_size, void* d_ws, size_t ws_size,
                              hipStream_t stream) {
    const float* x        = (const float*)d_in[0];
    // d_in[1] = mask (unused: causal mask computed analytically)
    const float* rope_sin = (const float*)d_in[2];
    const float* rope_cos = (const float*)d_in[3];
    const float* qkv_w    = (const float*)d_in[4];
    const float* qkv_b    = (const float*)d_in[5];
    const float* proj_w   = (const float*)d_in[6];
    const float* proj_b   = (const float*)d_in[7];
    float* out = (float*)d_out;
    (void)in_sizes; (void)n_in; (void)out_size; (void)ws_size;

    const size_t N_X     = (size_t)TOK * DIM;        // 4,194,304
    const size_t N_QKVW  = (size_t)DIM * 3 * DIM;    // 3,145,728
    const size_t N_PROJW = (size_t)DIM * DIM;        // 1,048,576
    const size_t N_QKV   = (size_t)TOK * 3 * DIM;    // 12,582,912

    char* ws = (char*)d_ws;
    size_t off = 0;
    auto take = [&](size_t bytes) -> char* {
        char* p = ws + off;
        off += (bytes + 255) & ~(size_t)255;
        return p;
    };
    bf16_t* xb     = (bf16_t*)take(N_X * 2);
    bf16_t* qkvwT  = (bf16_t*)take(N_QKVW * 2);
    bf16_t* projwT = (bf16_t*)take(N_PROJW * 2);
    float*  qkvf   = (float* )take(N_QKV * 4);
    bf16_t* Qb     = (bf16_t*)take(N_X * 2);
    bf16_t* Kb     = (bf16_t*)take(N_X * 2);
    bf16_t* VTb    = (bf16_t*)take(N_X * 2);
    bf16_t* Yb     = (bf16_t*)take(N_X * 2);

    f32_to_bf16_kernel<<<(int)((N_X + 255) / 256), 256, 0, stream>>>(x, xb, (int)N_X);
    transpose_w_kernel<<<dim3(DIM / 32, (3 * DIM) / 32), dim3(32, 8), 0, stream>>>(
        qkv_w, qkvwT, DIM, 3 * DIM);
    transpose_w_kernel<<<dim3(DIM / 32, DIM / 32), dim3(32, 8), 0, stream>>>(
        proj_w, projwT, DIM, DIM);

    gemm_bf16_kernel<<<dim3(TOK / 128, (3 * DIM) / 64), 128, 0, stream>>>(
        xb, qkvwT, qkv_b, qkvf, TOK, 3 * DIM, DIM);

    rope_qk_kernel<<<(int)((N_X + 255) / 256), 256, 0, stream>>>(
        qkvf, rope_sin, rope_cos, Qb, Kb);
    v_transpose_kernel<<<dim3(SEQ / 32, BATCH * NHEAD), 256, 0, stream>>>(qkvf, VTb);

    attn_kernel<<<dim3(SEQ / 16, BATCH * NHEAD), 32, 0, stream>>>(Qb, Kb, VTb, Yb);

    gemm_bf16_kernel<<<dim3(TOK / 128, DIM / 64), 128, 0, stream>>>(
        Yb, projwT, proj_b, out, TOK, DIM, DIM);
}
